// SelfAttentionFeedForward_13451837571471
// MI455X (gfx1250) — compile-verified, matrix operand order checked by reading, not run on hardware
//
#include <hip/hip_runtime.h>

// ---------------------------------------------------------------------------
// MI455X (gfx1250) implementation of SelfAttentionFeedForward.
//   * all 9 large GEMMs through v_wmma_f32_16x16x32_bf16 (fp32 accumulate)
//   * NT form C[m,n] = sum_k A[m,k]*B[n,k] for every GEMM -> one kernel
//   * global->LDS staging via GLOBAL_LOAD_ASYNC_TO_LDS_B128 (ASYNCcnt),
//     2-stage LDS double buffering, BK=64 (2 WMMA K-steps per stage)
// ---------------------------------------------------------------------------

typedef __bf16 bf16;
typedef __attribute__((ext_vector_type(8)))  bf16  v8bf;
typedef __attribute__((ext_vector_type(16))) bf16  v16bf;
typedef __attribute__((ext_vector_type(8)))  float v8f;
typedef __attribute__((ext_vector_type(4)))  float v4f;

namespace {
constexpr int cS = 8192;   // sequence
constexpr int cE = 512;    // embed
constexpr int cA = 2048;   // attention size
constexpr int cH = 4096;   // hidden
constexpr int cB = 295;    // broadcast bias quirk

constexpr int BM = 128;
constexpr int BN = 128;
constexpr int BK = 64;           // two WMMA K-steps per stage
constexpr int LDSS = BK + 8;     // padded LDS row stride (elements; 144B, 16B-aligned)
}

// ---------------------------- fp32 -> bf16 ---------------------------------
__global__ __launch_bounds__(256) void k_f32_to_bf16(const float* __restrict__ in,
                                                     bf16* __restrict__ out, int n) {
  int i = (blockIdx.x * 256 + threadIdx.x) * 4;
  if (i + 4 <= n) {
    v4f v = *(const v4f*)(in + i);
    out[i + 0] = (bf16)v.x;
    out[i + 1] = (bf16)v.y;
    out[i + 2] = (bf16)v.z;
    out[i + 3] = (bf16)v.w;
  } else {
    for (; i < n; ++i) out[i] = (bf16)in[i];
  }
}

// ---------------------- NT GEMM via bf16 WMMA ------------------------------
// C[m,n] = sum_k A[m,k] * B[n,k].  A: [M,K] bf16 row-major, B: [N,K] bf16
// row-major.  Output fp32 (Cf) or bf16 (Cb), optional fused bias+ReLU.
// 256 threads = 8 waves; block tile 128x128; wave tile 32x64.
// Async global->LDS copies, double-buffered.
template <bool OUT_BF16, bool FUSE_BIAS_RELU>
__global__ __launch_bounds__(256) void k_gemm_nt_wmma(
    const bf16* __restrict__ Am, const bf16* __restrict__ Bm,
    const float* __restrict__ bias,
    float* __restrict__ Cf, bf16* __restrict__ Cb,
    int M, int N, int K) {
  __shared__ bf16 As[2][BM * LDSS];
  __shared__ bf16 Bs[2][BN * LDSS];

  const int tid  = threadIdx.x;
  const int lane = tid & 31;
  const int wave = tid >> 5;      // 0..7
  const int wm   = wave >> 1;     // 0..3 : 32-row strip of the block tile
  const int wn   = wave & 1;      // 0..1 : 64-col strip

  const int bm0 = blockIdx.y * BM;
  const int bn0 = blockIdx.x * BN;

  v8f acc[2][4];
#pragma unroll
  for (int i = 0; i < 2; ++i)
#pragma unroll
    for (int j = 0; j < 4; ++j)
#pragma unroll
      for (int r = 0; r < 8; ++r) acc[i][j][r] = 0.0f;

  // Staging: each thread moves 64B of A and 64B of B per stage (4+4 B128s).
  const int lrow = tid >> 1;              // 0..127
  const int lcol = (tid & 1) * 32;        // 0 or 32 (bf16 elements)
  const bf16* aSrcBase = Am + (size_t)(bm0 + lrow) * K + lcol;
  const bf16* bSrcBase = Bm + (size_t)(bn0 + lrow) * K + lcol;
  const unsigned aLds[2] = {
      (unsigned)(unsigned long long)(uintptr_t)&As[0][lrow * LDSS + lcol],
      (unsigned)(unsigned long long)(uintptr_t)&As[1][lrow * LDSS + lcol]};
  const unsigned bLds[2] = {
      (unsigned)(unsigned long long)(uintptr_t)&Bs[0][lrow * LDSS + lcol],
      (unsigned)(unsigned long long)(uintptr_t)&Bs[1][lrow * LDSS + lcol]};

  const int kIters = K / BK;

  auto issue_stage = [&](int kt, int buf) {
    const bf16* as = aSrcBase + (size_t)kt * BK;
    const bf16* bs = bSrcBase + (size_t)kt * BK;
#pragma unroll
    for (int c = 0; c < 4; ++c) {
      asm volatile("global_load_async_to_lds_b128 %0, %1, off"
                   :: "v"(aLds[buf] + c * 16),
                      "v"((unsigned long long)(uintptr_t)(as + c * 8))
                   : "memory");
    }
#pragma unroll
    for (int c = 0; c < 4; ++c) {
      asm volatile("global_load_async_to_lds_b128 %0, %1, off"
                   :: "v"(bLds[buf] + c * 16),
                      "v"((unsigned long long)(uintptr_t)(bs + c * 8))
                   : "memory");
    }
  };

  // Fragment addressing (wave32 16-bit A-matrix 16x32 layout):
  // lane%16 -> matrix row; lane/16 selects K half: chunks [fk, fk+8) and
  // [fk+16, fk+24) within each 32-wide K-step.
  const int frow = lane & 15;
  const int fk   = (lane >> 4) * 8;

  issue_stage(0, 0);

  for (int kt = 0; kt < kIters; ++kt) {
    const int cur = kt & 1;
    if (kt + 1 < kIters) {
      issue_stage(kt + 1, cur ^ 1);
      // 8 ops of stage kt+1 outstanding; waiting to <=8 drains stage kt
      // (async loads return "done" in order).
      asm volatile("s_wait_asynccnt 0x8" ::: "memory");
    } else {
      asm volatile("s_wait_asynccnt 0x0" ::: "memory");
    }
    __syncthreads();   // stage kt visible to all waves

#pragma unroll
    for (int ks = 0; ks < 2; ++ks) {
      const int kbase = ks * 32 + fk;
      v16bf afrag[2], bfrag[4];
#pragma unroll
      for (int i = 0; i < 2; ++i) {
        const bf16* p = &As[cur][(wm * 32 + i * 16 + frow) * LDSS + kbase];
        v8bf lo = *(const v8bf*)(p);
        v8bf hi = *(const v8bf*)(p + 16);
        afrag[i] = __builtin_shufflevector(lo, hi, 0, 1, 2, 3, 4, 5, 6, 7,
                                           8, 9, 10, 11, 12, 13, 14, 15);
      }
#pragma unroll
      for (int j = 0; j < 4; ++j) {
        const bf16* p = &Bs[cur][(wn * 64 + j * 16 + frow) * LDSS + kbase];
        v8bf lo = *(const v8bf*)(p);
        v8bf hi = *(const v8bf*)(p + 16);
        bfrag[j] = __builtin_shufflevector(lo, hi, 0, 1, 2, 3, 4, 5, 6, 7,
                                           8, 9, 10, 11, 12, 13, 14, 15);
      }
#pragma unroll
      for (int i = 0; i < 2; ++i)
#pragma unroll
        for (int j = 0; j < 4; ++j)
          acc[i][j] = __builtin_amdgcn_wmma_f32_16x16x32_bf16(
              false, afrag[i], false, bfrag[j], (short)0, acc[i][j], false, false);
    }

    __syncthreads();   // all waves done reading buf `cur`; it may be refilled
  }

  // Epilogue: C/D 16x16 f32 layout -> lane owns col n = lane%16,
  // rows (lane/16)*8 .. +7 in acc elements 0..7.
  const int rbase = (lane >> 4) * 8;
#pragma unroll
  for (int i = 0; i < 2; ++i) {
#pragma unroll
    for (int j = 0; j < 4; ++j) {
      const int gm = bm0 + wm * 32 + i * 16 + rbase;
      const int gn = bn0 + wn * 64 + j * 16 + frow;
      float bv = 0.0f;
      if (FUSE_BIAS_RELU) bv = bias[gn];
#pragma unroll
      for (int r = 0; r < 8; ++r) {
        float v = acc[i][j][r];
        if (FUSE_BIAS_RELU) {
          v += bv;
          v = v > 0.0f ? v : 0.0f;
        }
        const size_t idx = (size_t)(gm + r) * (size_t)N + (size_t)gn;
        if (OUT_BF16) Cb[idx] = (bf16)v;
        else          Cf[idx] = v;
      }
    }
  }
}

// -------------------- softmax over axis 0 (per column) ---------------------
__global__ __launch_bounds__(256) void k_softmax_axis0(const float* __restrict__ att,
                                                       bf16* __restrict__ w, int n) {
  const int b = blockIdx.x * 256 + threadIdx.x;
  if (b >= n) return;
  float mx = -3.4e38f;
  for (int a = 0; a < n; ++a) mx = fmaxf(mx, att[(size_t)a * n + b]);
  float sum = 0.0f;
  for (int a = 0; a < n; ++a) sum += __expf(att[(size_t)a * n + b] - mx);
  const float inv = 1.0f / sum;
  for (int a = 0; a < n; ++a)
    w[(size_t)a * n + b] = (bf16)(__expf(att[(size_t)a * n + b] - mx) * inv);
}

// ------------------- final H->1 projection (per row dot) -------------------
__global__ __launch_bounds__(256) void k_rowdot(const bf16* __restrict__ Hm,
                                                const bf16* __restrict__ wout,
                                                const float* __restrict__ bout,
                                                float* __restrict__ out, int ncols) {
  __shared__ float red[256];
  const int s = blockIdx.x;
  float sum = 0.0f;
  for (int h = threadIdx.x; h < ncols; h += 256)
    sum += (float)Hm[(size_t)s * ncols + h] * (float)wout[h];
  red[threadIdx.x] = sum;
  __syncthreads();
  for (int off = 128; off > 0; off >>= 1) {
    if (threadIdx.x < off) red[threadIdx.x] += red[threadIdx.x + off];
    __syncthreads();
  }
  if (threadIdx.x == 0) out[s] = red[0] + bout[0];
}

// -------- faithful quirk: sigmoid([S,1] + b[295]) broadcast to [S,295] -----
__global__ __launch_bounds__(256) void k_bcast_sigmoid(const float* __restrict__ rowout,
                                                       const float* __restrict__ b,
                                                       float* __restrict__ out,
                                                       int total, int nb) {
  const int i = blockIdx.x * 256 + threadIdx.x;
  if (i >= total) return;
  const int s = i / nb;
  const int j = i - s * nb;
  const float v = rowout[s] + b[j];
  out[i] = 1.0f / (1.0f + __expf(-v));
}

// ---------------------------------------------------------------------------
extern "C" void kernel_launch(void* const* d_in, const int* in_sizes, int n_in,
                              void* d_out, int out_size, void* d_ws, size_t ws_size,
                              hipStream_t stream) {
  (void)in_sizes; (void)n_in; (void)out_size; (void)ws_size;

  const float* x    = (const float*)d_in[0];
  const float* W_Q  = (const float*)d_in[1];
  const float* W_K  = (const float*)d_in[2];
  const float* W_V  = (const float*)d_in[3];
  const float* bvec = (const float*)d_in[4];
  const float* W1   = (const float*)d_in[5];
  const float* b1   = (const float*)d_in[6];
  const float* W2   = (const float*)d_in[7];
  const float* b2   = (const float*)d_in[8];
  const float* W3   = (const float*)d_in[9];
  const float* b3   = (const float*)d_in[10];
  const float* W4   = (const float*)d_in[11];
  const float* b4   = (const float*)d_in[12];
  const float* Wout = (const float*)d_in[13];
  const float* bout = (const float*)d_in[14];
  float* out = (float*)d_out;

  char* ws = (char*)d_ws;
  size_t off = 0;
  auto alloc = [&](size_t bytes) -> void* {
    off = (off + 255) & ~(size_t)255;
    void* p = ws + off;
    off += bytes;
    return p;
  };

  bf16*  x_bf    = (bf16*)alloc((size_t)cS * cE * 2);
  bf16*  wq_bf   = (bf16*)alloc((size_t)cA * cE * 2);
  bf16*  wk_bf   = (bf16*)alloc((size_t)cA * cE * 2);
  bf16*  wv_bf   = (bf16*)alloc((size_t)cA * cE * 2);
  bf16*  q_bf    = (bf16*)alloc((size_t)cA * cS * 2);   // [A,S]
  bf16*  k_bf    = (bf16*)alloc((size_t)cA * cS * 2);   // [A,S]
  bf16*  vT_bf   = (bf16*)alloc((size_t)cS * cA * 2);   // [S,A]
  float* att_f   = (float*)alloc((size_t)cA * cA * 4);  // [A,A]
  bf16*  w_bf    = (bf16*)alloc((size_t)cA * cA * 2);   // softmax(att)
  bf16*  ctx_bf  = (bf16*)alloc((size_t)cS * cA * 2);   // [S,A]
  bf16*  wbig    = (bf16*)alloc((size_t)cH * cH * 2);   // reused W1..W4
  bf16*  hA      = (bf16*)alloc((size_t)cS * cH * 2);
  bf16*  hB      = (bf16*)alloc((size_t)cS * cH * 2);
  bf16*  wout_bf = (bf16*)alloc((size_t)cH * 2);
  float* rowout  = (float*)alloc((size_t)cS * 4);

  auto cvt = [&](const float* src, bf16* dst, int n) {
    int blocks = ((n + 3) / 4 + 255) / 256;
    k_f32_to_bf16<<<blocks, 256, 0, stream>>>(src, dst, n);
  };

  // Stage 0: convert activations + QKV weights.
  cvt(x,   x_bf,  cS * cE);
  cvt(W_Q, wq_bf, cA * cE);
  cvt(W_K, wk_bf, cA * cE);
  cvt(W_V, wv_bf, cA * cE);

  // q[a,s] = sum_e W_Q[a,e] x[s,e]   (M=A, N=S, K=E)
  k_gemm_nt_wmma<true, false><<<dim3(cS / BN, cA / BM), 256, 0, stream>>>(
      wq_bf, x_bf, nullptr, nullptr, q_bf, cA, cS, cE);
  // k[a,s]
  k_gemm_nt_wmma<true, false><<<dim3(cS / BN, cA / BM), 256, 0, stream>>>(
      wk_bf, x_bf, nullptr, nullptr, k_bf, cA, cS, cE);
  // vT[s,a] = sum_e x[s,e] W_V[a,e]  (M=S, N=A, K=E)
  k_gemm_nt_wmma<true, false><<<dim3(cA / BN, cS / BM), 256, 0, stream>>>(
      x_bf, wv_bf, nullptr, nullptr, vT_bf, cS, cA, cE);

  // att[a,b] = sum_s q[a,s] k[b,s]   (M=A, N=A, K=S) -> fp32 for softmax
  k_gemm_nt_wmma<false, false><<<dim3(cA / BN, cA / BM), 256, 0, stream>>>(
      q_bf, k_bf, nullptr, att_f, nullptr, cA, cA, cS);

  // column softmax (axis 0)
  k_softmax_axis0<<<cA / 256, 256, 0, stream>>>(att_f, w_bf, cA);

  // ctx[s,a] = sum_b vT[s,b] w[a,b]  (M=S, N=A, K=A)
  k_gemm_nt_wmma<true, false><<<dim3(cA / BN, cS / BM), 256, 0, stream>>>(
      vT_bf, w_bf, nullptr, nullptr, ctx_bf, cS, cA, cA);

  // MLP: h1 = relu(ctx @ W1^T + b1)  (M=S, N=H, K=A)
  cvt(W1, wbig, cH * cA);
  k_gemm_nt_wmma<true, true><<<dim3(cH / BN, cS / BM), 256, 0, stream>>>(
      ctx_bf, wbig, b1, nullptr, hA, cS, cH, cA);
  // h2..h4  (M=S, N=H, K=H)
  cvt(W2, wbig, cH * cH);
  k_gemm_nt_wmma<true, true><<<dim3(cH / BN, cS / BM), 256, 0, stream>>>(
      hA, wbig, b2, nullptr, hB, cS, cH, cH);
  cvt(W3, wbig, cH * cH);
  k_gemm_nt_wmma<true, true><<<dim3(cH / BN, cS / BM), 256, 0, stream>>>(
      hB, wbig, b3, nullptr, hA, cS, cH, cH);
  cvt(W4, wbig, cH * cH);
  k_gemm_nt_wmma<true, true><<<dim3(cH / BN, cS / BM), 256, 0, stream>>>(
      hA, wbig, b4, nullptr, hB, cS, cH, cH);

  // out[s] = h4[s,:] . Wout + bout
  cvt(Wout, wout_bf, cH);
  k_rowdot<<<cS, 256, 0, stream>>>(hB, wout_bf, bout, rowout, cH);

  // sigmoid([S,1] + b[295]) -> [S,295]
  const int total = cS * cB;
  k_bcast_sigmoid<<<(total + 255) / 256, 256, 0, stream>>>(rowout, bvec, out, total, cB);
}
